// FeaturePropagation_446676599134
// MI455X (gfx1250) — matrix operand order repaired and checked by reading, not compiled, not run on hardware
//
#include <hip/hip_runtime.h>

// ---------------------------------------------------------------------------
// PointNet++ FeaturePropagation for MI455X (gfx1250, wave32, WMMA).
// GEMMs in bf16 (v_wmma_f32_16x16x32_bf16, f32 accumulate); everything
// numerically sensitive (distances, 1/d weights, BN scale/bias) stays f32.
// Weight staging to LDS uses gfx1250 async-to-LDS loads (ASYNCcnt).
// ---------------------------------------------------------------------------

#define BN_EPS 1e-5f
constexpr int kB  = 4;
constexpr int kN1 = 16384;
constexpr int kN2 = 4096;
constexpr int kC  = 128;
constexpr int CH0 = 3 * kC;   // 384 = GEMM0 K
constexpr int OC  = 256;      // output channels both layers

typedef __bf16 bf16;
typedef __attribute__((ext_vector_type(16))) __bf16 v16bf;
typedef __attribute__((ext_vector_type(8)))  __bf16 v8bf;
typedef __attribute__((ext_vector_type(8)))  float  v8f;

union Frag  { v16bf v; v8bf h[2]; };
union Pack8 { v8bf  v; uint4 u;   };

// Async bulk copy: global -> LDS, 16B per lane, no VGPR round-trip.
// lds_off is the wave-relative LDS byte offset (low 32 bits of a generic LDS
// pointer, per the flat-aperture truncation rule).
__device__ __forceinline__ void async_b128_to_lds(unsigned lds_off, const void* gaddr) {
  asm volatile("global_load_async_to_lds_b128 %0, %1, off"
               :: "v"(lds_off), "v"(gaddr) : "memory");
}
__device__ __forceinline__ void wait_asynccnt0() {
  asm volatile("s_wait_asynccnt 0x0" ::: "memory");
}

// ---------------------------------------------------------------------------
// Workspace layout (all offsets 256B aligned), total ~94.2 MB
// ---------------------------------------------------------------------------
constexpr size_t OFF_W0BF = 0;                         // 256*384 bf16  = 196608 B
constexpr size_t OFF_W1BF = 196608;                    // 256*256 bf16  = 131072 B
constexpr size_t OFF_SCL  = 327680;                    // 4*256 f32     =   4096 B
constexpr size_t OFF_P2T  = 331776;                    // B*N2*C f32    = 8388608 B
constexpr size_t OFF_IDX  = 8720384;                   // B*N1*3 i32    = 786432 B
constexpr size_t OFF_WGT  = 9506816;                   // B*N1*3 f32    = 786432 B
constexpr size_t OFF_XT   = 10293248;                  // B*N1*384 bf16 = 50331648 B
constexpr size_t OFF_Y0T  = 60624896;                  // B*N1*256 bf16 = 33554432 B

// ---------------------------------------------------------------------------
// 1) Weight conversion + BN folding:  y = s*(W x) + t,  relu
//    s = g/sqrt(v+eps); t = beta + (b - m)*s
// ---------------------------------------------------------------------------
__global__ void prep_kernel(const float* __restrict__ w0, const float* __restrict__ w1,
                            const float* __restrict__ b0, const float* __restrict__ g0,
                            const float* __restrict__ bt0, const float* __restrict__ m0,
                            const float* __restrict__ v0,
                            const float* __restrict__ b1, const float* __restrict__ g1,
                            const float* __restrict__ bt1, const float* __restrict__ m1,
                            const float* __restrict__ v1,
                            bf16* __restrict__ w0bf, bf16* __restrict__ w1bf,
                            float* __restrict__ scl) {
  int tid = blockIdx.x * blockDim.x + threadIdx.x;
  if (tid < OC * CH0) w0bf[tid] = (bf16)w0[tid];
  int t1 = tid - OC * CH0;
  if (t1 >= 0 && t1 < OC * OC) w1bf[t1] = (bf16)w1[t1];
  if (blockIdx.x == 0 && threadIdx.x < OC) {
    int c = threadIdx.x;
    float s0 = g0[c] * rsqrtf(v0[c] + BN_EPS);
    scl[c]       = s0;
    scl[256 + c] = bt0[c] + (b0[c] - m0[c]) * s0;
    float s1 = g1[c] * rsqrtf(v1[c] + BN_EPS);
    scl[512 + c] = s1;
    scl[768 + c] = bt1[c] + (b1[c] - m1[c]) * s1;
  }
}

// ---------------------------------------------------------------------------
// 2) points2 (B,C,N2) f32 -> p2t (B,N2,C) f32  (coalesced gather source)
// ---------------------------------------------------------------------------
__global__ void transpose_p2_kernel(const float* __restrict__ p2, float* __restrict__ p2t) {
  __shared__ float tile[32][33];
  int b = blockIdx.z, n0 = blockIdx.x * 32, c0 = blockIdx.y * 32;
  int tx = threadIdx.x, ty = threadIdx.y;
#pragma unroll
  for (int i = 0; i < 4; i++)
    tile[ty + 8 * i][tx] = p2[((size_t)b * kC + (c0 + ty + 8 * i)) * kN2 + n0 + tx];
  __syncthreads();
#pragma unroll
  for (int i = 0; i < 4; i++)
    p2t[((size_t)b * kN2 + n0 + ty + 8 * i) * kC + c0 + tx] = tile[tx][ty + 8 * i];
}

// ---------------------------------------------------------------------------
// 3) 3-NN: one thread per query point; xyz2 tiled through LDS.
// ---------------------------------------------------------------------------
__global__ void nn3_kernel(const float* __restrict__ xyz1, const float* __restrict__ xyz2,
                           int* __restrict__ idx3, float* __restrict__ wgt3) {
  constexpr int TS = 1024;
  __shared__ float s[TS * 3];
  int b = blockIdx.x >> 6;                        // 64 blocks per batch
  int n = ((blockIdx.x & 63) << 8) + threadIdx.x;
  const float* p = xyz1 + ((size_t)b * kN1 + n) * 3;
  float x = p[0], y = p[1], z = p[2];
  float d0 = 3.4e38f, d1 = 3.4e38f, d2 = 3.4e38f;
  int   i0 = 0, i1 = 0, i2 = 0;
  for (int t0 = 0; t0 < kN2; t0 += TS) {
    const float* q = xyz2 + ((size_t)b * kN2 + t0) * 3;
    for (int k = threadIdx.x; k < TS * 3; k += 256) s[k] = q[k];
    __syncthreads();
    for (int j = 0; j < TS; j++) {               // LDS broadcast reads
      float dx = x - s[3 * j], dy = y - s[3 * j + 1], dz = z - s[3 * j + 2];
      float d = dx * dx + dy * dy + dz * dz;
      int jj = t0 + j;
      if (d < d2) {
        if (d < d1) {
          if (d < d0) { d2 = d1; i2 = i1; d1 = d0; i1 = i0; d0 = d; i0 = jj; }
          else        { d2 = d1; i2 = i1; d1 = d;  i1 = jj; }
        } else        { d2 = d;  i2 = jj; }
      }
    }
    __syncthreads();
  }
  float w0 = 1.0f / fmaxf(d0, 1e-10f);
  float w1 = 1.0f / fmaxf(d1, 1e-10f);
  float w2 = 1.0f / fmaxf(d2, 1e-10f);
  float inv = 1.0f / (w0 + w1 + w2);
  size_t o = ((size_t)b * kN1 + n) * 3;
  idx3[o] = i0; idx3[o + 1] = i1; idx3[o + 2] = i2;
  wgt3[o] = w0 * inv; wgt3[o + 1] = w1 * inv; wgt3[o + 2] = w2 * inv;
}

// ---------------------------------------------------------------------------
// 4) src (B,C,N1) f32 -> Xt rows [coff, coff+128) bf16 (point-major)
// ---------------------------------------------------------------------------
__global__ void transpose_to_xt_kernel(const float* __restrict__ src, bf16* __restrict__ xt,
                                       int coff) {
  __shared__ float tile[32][33];
  int b = blockIdx.z, n0 = blockIdx.x * 32, c0 = blockIdx.y * 32;
  int tx = threadIdx.x, ty = threadIdx.y;
#pragma unroll
  for (int i = 0; i < 4; i++)
    tile[ty + 8 * i][tx] = src[((size_t)b * kC + (c0 + ty + 8 * i)) * kN1 + n0 + tx];
  __syncthreads();
#pragma unroll
  for (int i = 0; i < 4; i++)
    xt[((size_t)(b * kN1 + n0 + ty + 8 * i)) * CH0 + coff + c0 + tx] =
        (bf16)tile[tx][ty + 8 * i];
}

// ---------------------------------------------------------------------------
// 5) Interpolation: one warp per point, 4 channels/lane, coalesced float4.
//    Writes Xt rows [128,256) bf16.
// ---------------------------------------------------------------------------
__global__ void interp_kernel(const float* __restrict__ p2t, const int* __restrict__ idx3,
                              const float* __restrict__ wgt3, bf16* __restrict__ xt) {
  int lane = threadIdx.x & 31;
  int w    = threadIdx.x >> 5;
  int p    = blockIdx.x * 8 + w;                 // flat point in [0, B*N1)
  int b    = p >> 14;                            // N1 = 16384
  size_t o3 = (size_t)p * 3;
  int i0 = idx3[o3], i1 = idx3[o3 + 1], i2 = idx3[o3 + 2];
  float w0 = wgt3[o3], w1 = wgt3[o3 + 1], w2 = wgt3[o3 + 2];
  const float4* r0 = (const float4*)(p2t + ((size_t)b * kN2 + i0) * kC);
  const float4* r1 = (const float4*)(p2t + ((size_t)b * kN2 + i1) * kC);
  const float4* r2 = (const float4*)(p2t + ((size_t)b * kN2 + i2) * kC);
  float4 a = r0[lane], c = r1[lane], d = r2[lane];
  float fx = w0 * a.x + w1 * c.x + w2 * d.x;
  float fy = w0 * a.y + w1 * c.y + w2 * d.y;
  float fz = w0 * a.z + w1 * c.z + w2 * d.z;
  float fw = w0 * a.w + w1 * c.w + w2 * d.w;
  union { bf16 h[4]; uint2 u; } pk;
  pk.h[0] = (bf16)fx; pk.h[1] = (bf16)fy; pk.h[2] = (bf16)fz; pk.h[3] = (bf16)fw;
  *(uint2*)(xt + (size_t)p * CH0 + kC + lane * 4) = pk.u;
}

// ---------------------------------------------------------------------------
// 6) GEMM0: Y0 = relu(s0*(W0 @ X) + t0)  via v_wmma_f32_16x16x32_bf16.
//    Block = 256 threads (8 waves); each wave owns TWO 16-point N-tiles so
//    every A-fragment LDS load feeds two WMMAs. 24 resident B fragments need
//    ~236 VGPRs -> __launch_bounds__(256,1) lifts the occupancy-driven VGPR
//    cap (LDS already limits the WGP to this one workgroup, so high VGPR use
//    is free). W0 (bf16) staged in LDS via async-to-LDS with row pitch 392
//    bf16 (784 B, 784%256==16 -> 16 lane rows hit all 64 banks, conflict-free
//    ds_load_b128).
// ---------------------------------------------------------------------------
__global__ void __launch_bounds__(256, 1)
gemm0_kernel(const bf16* __restrict__ w0bf, const float* __restrict__ scl,
             const bf16* __restrict__ xt, bf16* __restrict__ y0t) {
  extern __shared__ char smem[];
  constexpr int PITCH = 392;                      // bf16 elems / LDS row
  bf16*  wl = (bf16*)smem;                        // 256*392*2 = 200704 B
  float* s0 = (float*)(smem + OC * PITCH * 2);
  float* t0 = s0 + OC;
  {
    int r = threadIdx.x;                          // 256 threads == 256 rows
    const char* src = (const char*)(w0bf + (size_t)r * CH0);
    unsigned    dst = (unsigned)(size_t)(wl + (size_t)r * PITCH);
#pragma unroll
    for (int i = 0; i < CH0 * 2 / 16; i++)        // 48 x async b128 per row
      async_b128_to_lds(dst + i * 16, src + i * 16);
    s0[r] = scl[r];
    t0[r] = scl[256 + r];
    wait_asynccnt0();
  }
  __syncthreads();
  int wave = threadIdx.x >> 5, lane = threadIdx.x & 31;
  int half = lane >> 4, nl = lane & 15;
  size_t p0 = (size_t)blockIdx.x * 256 + wave * 32 + nl;      // flat point rows
  // B fragments (32x16 bf16): elems 0-7 = K half*8..+7, elems 8-15 = K 16+half*8..+7
  Frag bfr[2][12];
  const bf16* xp0 = xt + p0 * CH0;
  const bf16* xp1 = xp0 + (size_t)16 * CH0;
#pragma unroll
  for (int ks = 0; ks < 12; ks++) {
    const v8bf* q0 = (const v8bf*)(xp0 + ks * 32 + half * 8);
    const v8bf* q1 = (const v8bf*)(xp1 + ks * 32 + half * 8);
    bfr[0][ks].h[0] = q0[0]; bfr[0][ks].h[1] = q0[2];
    bfr[1][ks].h[0] = q1[0]; bfr[1][ks].h[1] = q1[2];
  }
  bf16* yp0 = y0t + p0 * OC;
  bf16* yp1 = yp0 + (size_t)16 * OC;
  for (int mt = 0; mt < 16; mt++) {
    v8f acc0, acc1;
#pragma unroll
    for (int r = 0; r < 8; r++) { acc0[r] = 0.0f; acc1[r] = 0.0f; }
    const bf16* wrow = wl + (size_t)(mt * 16 + nl) * PITCH;
#pragma unroll
    for (int ks = 0; ks < 12; ks++) {
      // A frag (16x32 bf16): elems 0-7 = K half*8..+7, elems 8-15 = K 16+half*8..+7
      Frag a;
      a.h[0] = *(const v8bf*)(wrow + ks * 32 + half * 8);
      a.h[1] = *(const v8bf*)(wrow + ks * 32 + 16 + half * 8);
      acc0 = __builtin_amdgcn_wmma_f32_16x16x32_bf16(false, a.v, false, bfr[0][ks].v,
                                                     (short)0, acc0, false, false);
      acc1 = __builtin_amdgcn_wmma_f32_16x16x32_bf16(false, a.v, false, bfr[1][ks].v,
                                                     (short)0, acc1, false, false);
    }
    int cbase = mt * 16 + half * 8;               // C/D: VGPR r -> M = cbase + r
    Pack8 o0, o1;
#pragma unroll
    for (int r = 0; r < 8; r++) {
      float sc = s0[cbase + r], tc = t0[cbase + r];
      o0.v[r] = (bf16)fmaxf(acc0[r] * sc + tc, 0.0f);
      o1.v[r] = (bf16)fmaxf(acc1[r] * sc + tc, 0.0f);
    }
    *(uint4*)(yp0 + cbase) = o0.u;                // 16B store per lane
    *(uint4*)(yp1 + cbase) = o1.u;
  }
}

// ---------------------------------------------------------------------------
// 7) GEMM1 + channel-max: out[p] = max_c relu(s1*(W1 @ Y0) + t1)
//    Same two-N-tile scheme; LDS pitch 264 bf16 (528 B, 528%256==16).
// ---------------------------------------------------------------------------
__global__ void __launch_bounds__(256, 1)
gemm1_kernel(const bf16* __restrict__ w1bf, const float* __restrict__ scl,
             const bf16* __restrict__ y0t, float* __restrict__ out) {
  extern __shared__ char smem[];
  constexpr int PITCH = 264;
  bf16*  wl = (bf16*)smem;                        // 256*264*2 = 135168 B
  float* s1 = (float*)(smem + OC * PITCH * 2);
  float* t1 = s1 + OC;
  {
    int r = threadIdx.x;
    const char* src = (const char*)(w1bf + (size_t)r * OC);
    unsigned    dst = (unsigned)(size_t)(wl + (size_t)r * PITCH);
#pragma unroll
    for (int i = 0; i < OC * 2 / 16; i++)         // 32 x async b128 per row
      async_b128_to_lds(dst + i * 16, src + i * 16);
    s1[r] = scl[512 + r];
    t1[r] = scl[768 + r];
    wait_asynccnt0();
  }
  __syncthreads();
  int wave = threadIdx.x >> 5, lane = threadIdx.x & 31;
  int half = lane >> 4, nl = lane & 15;
  size_t p0 = (size_t)blockIdx.x * 256 + wave * 32 + nl;
  Frag bfr[2][8];
  const bf16* yp0 = y0t + p0 * OC;
  const bf16* yp1 = yp0 + (size_t)16 * OC;
#pragma unroll
  for (int ks = 0; ks < 8; ks++) {
    const v8bf* q0 = (const v8bf*)(yp0 + ks * 32 + half * 8);
    const v8bf* q1 = (const v8bf*)(yp1 + ks * 32 + half * 8);
    bfr[0][ks].h[0] = q0[0]; bfr[0][ks].h[1] = q0[2];
    bfr[1][ks].h[0] = q1[0]; bfr[1][ks].h[1] = q1[2];
  }
  float vmax0 = -3.4e38f, vmax1 = -3.4e38f;
  for (int mt = 0; mt < 16; mt++) {
    v8f acc0, acc1;
#pragma unroll
    for (int r = 0; r < 8; r++) { acc0[r] = 0.0f; acc1[r] = 0.0f; }
    const bf16* wrow = wl + (size_t)(mt * 16 + nl) * PITCH;
#pragma unroll
    for (int ks = 0; ks < 8; ks++) {
      Frag a;
      a.h[0] = *(const v8bf*)(wrow + ks * 32 + half * 8);
      a.h[1] = *(const v8bf*)(wrow + ks * 32 + 16 + half * 8);
      acc0 = __builtin_amdgcn_wmma_f32_16x16x32_bf16(false, a.v, false, bfr[0][ks].v,
                                                     (short)0, acc0, false, false);
      acc1 = __builtin_amdgcn_wmma_f32_16x16x32_bf16(false, a.v, false, bfr[1][ks].v,
                                                     (short)0, acc1, false, false);
    }
    int cbase = mt * 16 + half * 8;
#pragma unroll
    for (int r = 0; r < 8; r++) {
      float sc = s1[cbase + r], tc = t1[cbase + r];
      vmax0 = fmaxf(vmax0, fmaxf(acc0[r] * sc + tc, 0.0f));
      vmax1 = fmaxf(vmax1, fmaxf(acc1[r] * sc + tc, 0.0f));
    }
  }
  // lanes L and L+16 hold complementary channel halves of the same point
  vmax0 = fmaxf(vmax0, __shfl_xor(vmax0, 16, 32));
  vmax1 = fmaxf(vmax1, __shfl_xor(vmax1, 16, 32));
  if (half == 0) {
    out[p0] = vmax0;
    out[p0 + 16] = vmax1;
  }
}

// ---------------------------------------------------------------------------
extern "C" void kernel_launch(void* const* d_in, const int* in_sizes, int n_in,
                              void* d_out, int out_size, void* d_ws, size_t ws_size,
                              hipStream_t stream) {
  const float* xyz1      = (const float*)d_in[0];
  const float* xyz2      = (const float*)d_in[1];
  const float* points2   = (const float*)d_in[2];
  const float* points1   = (const float*)d_in[3];
  const float* points_b1 = (const float*)d_in[4];
  const float* w0 = (const float*)d_in[5];
  const float* b0 = (const float*)d_in[6];
  const float* g0 = (const float*)d_in[7];
  const float* bt0 = (const float*)d_in[8];
  const float* m0 = (const float*)d_in[9];
  const float* v0 = (const float*)d_in[10];
  const float* w1 = (const float*)d_in[11];
  const float* b1 = (const float*)d_in[12];
  const float* g1 = (const float*)d_in[13];
  const float* bt1 = (const float*)d_in[14];
  const float* m1 = (const float*)d_in[15];
  const float* v1 = (const float*)d_in[16];

  char* ws   = (char*)d_ws;
  bf16*  w0bf = (bf16*)(ws + OFF_W0BF);
  bf16*  w1bf = (bf16*)(ws + OFF_W1BF);
  float* scl  = (float*)(ws + OFF_SCL);
  float* p2t  = (float*)(ws + OFF_P2T);
  int*   idx3 = (int*)(ws + OFF_IDX);
  float* wgt3 = (float*)(ws + OFF_WGT);
  bf16*  xt   = (bf16*)(ws + OFF_XT);
  bf16*  y0t  = (bf16*)(ws + OFF_Y0T);
  float* outp = (float*)d_out;

  // allow >64KB dynamic LDS for the GEMM kernels (WGP has 320KB)
  hipFuncSetAttribute(reinterpret_cast<const void*>(gemm0_kernel),
                      hipFuncAttributeMaxDynamicSharedMemorySize, 202752);
  hipFuncSetAttribute(reinterpret_cast<const void*>(gemm1_kernel),
                      hipFuncAttributeMaxDynamicSharedMemorySize, 137216);

  prep_kernel<<<640, 256, 0, stream>>>(w0, w1, b0, g0, bt0, m0, v0,
                                       b1, g1, bt1, m1, v1, w0bf, w1bf, scl);
  transpose_p2_kernel<<<dim3(kN2 / 32, kC / 32, kB), dim3(32, 8), 0, stream>>>(points2, p2t);
  nn3_kernel<<<kB * kN1 / 256, 256, 0, stream>>>(xyz1, xyz2, idx3, wgt3);
  transpose_to_xt_kernel<<<dim3(kN1 / 32, kC / 32, kB), dim3(32, 8), 0, stream>>>(points1, xt, 0);
  transpose_to_xt_kernel<<<dim3(kN1 / 32, kC / 32, kB), dim3(32, 8), 0, stream>>>(points_b1, xt, 2 * kC);
  interp_kernel<<<kB * kN1 / 8, 256, 0, stream>>>(p2t, idx3, wgt3, xt);
  gemm0_kernel<<<kB * kN1 / 256, 256, 202752, stream>>>(w0bf, scl, xt, y0t);
  gemm1_kernel<<<kB * kN1 / 256, 256, 137216, stream>>>(w1bf, scl, y0t, outp);
}
